// ToggleHeteroConvGNN_90855738180232
// MI455X (gfx1250) — matrix operand order repaired and checked by reading, not compiled, not run on hardware
//
#include <hip/hip_runtime.h>
#include <hip/hip_bf16.h>

typedef __attribute__((ext_vector_type(16))) __bf16 v16bf;
typedef __attribute__((ext_vector_type(8)))  float  v8f;
typedef __attribute__((ext_vector_type(4)))  unsigned int u32x4;
typedef __attribute__((ext_vector_type(8)))  unsigned int u32x8;

#define HFEAT 128

// ---------------------------------------------------------------- utilities

__global__ void zero_f32_kernel(float* __restrict__ p, long long n) {
  long long i = (long long)blockIdx.x * blockDim.x + threadIdx.x;
  long long stride = (long long)gridDim.x * blockDim.x;
  for (; i < n; i += stride) p[i] = 0.0f;
}

__global__ void cvt_bf16_kernel(const float* __restrict__ s, __bf16* __restrict__ d, long long n) {
  long long i = (long long)blockIdx.x * blockDim.x + threadIdx.x;
  long long stride = (long long)gridDim.x * blockDim.x;
  for (; i < n; i += stride) d[i] = (__bf16)s[i];
}

__global__ void count_kernel(const int* __restrict__ dst, float* __restrict__ cnt, int E) {
  int i = blockIdx.x * blockDim.x + threadIdx.x;
  int stride = gridDim.x * blockDim.x;
  for (; i < E; i += stride) unsafeAtomicAdd(&cnt[dst[i]], 1.0f);
}

__global__ void inv_kernel(float* __restrict__ c, long long n) {
  long long i = (long long)blockIdx.x * blockDim.x + threadIdx.x;
  long long stride = (long long)gridDim.x * blockDim.x;
  for (; i < n; i += stride) c[i] = 1.0f / fmaxf(c[i], 1.0f);
}

__global__ void update_kernel(float* __restrict__ h, const float* __restrict__ nw, long long n) {
  long long i = (long long)blockIdx.x * blockDim.x + threadIdx.x;
  long long stride = (long long)gridDim.x * blockDim.x;
  for (; i < n; i += stride) h[i] = fmaxf(h[i] + nw[i], 0.0f);
}

// ------------------------------------------------- edge scatter (mean-sum)
// One wave per edge: lane loads float4 of the 128-float source row and does
// four f32 atomic adds into the (L2-resident) destination accumulator.

__global__ void scatter_kernel(const float* __restrict__ xsrc, const int* __restrict__ src,
                               const int* __restrict__ dst, float* __restrict__ acc, int E) {
  int warp = (blockIdx.x * blockDim.x + threadIdx.x) >> 5;
  int lane = threadIdx.x & 31;
  int nwarp = (gridDim.x * blockDim.x) >> 5;
  for (int e = warp; e < E; e += nwarp) {
    int s = src[e];
    int d = dst[e];
    float4 v = ((const float4*)(xsrc + (long long)s * HFEAT))[lane];
    float* p = acc + (long long)d * HFEAT + lane * 4;
    unsafeAtomicAdd(p + 0, v.x);
    unsafeAtomicAdd(p + 1, v.y);
    unsafeAtomicAdd(p + 2, v.z);
    unsafeAtomicAdd(p + 3, v.w);
  }
}

// ----------------------------------------------- small-K dense encoders

__global__ void encode_small_kernel(const float* __restrict__ x, const float* __restrict__ W,
                                    const float* __restrict__ b, float* __restrict__ out,
                                    long long N, int Kin, int ldx, int xoff, int ldo, int ooff) {
  long long total = N * HFEAT;
  long long i = (long long)blockIdx.x * blockDim.x + threadIdx.x;
  long long stride = (long long)gridDim.x * blockDim.x;
  for (; i < total; i += stride) {
    long long n = i >> 7;
    int h = (int)(i & 127);
    const float* xr = x + n * ldx + xoff;
    const float* wr = W + (long long)h * Kin;
    float s = b[h];
    for (int k = 0; k < Kin; ++k) s += xr[k] * wr[k];
    out[n * ldo + ooff + h] = fmaxf(s, 0.0f);
  }
}

// ------------------------------------------------- TDM: stage 2D f32 tile to LDS
// Packs a D# (group0: 4 SGPRs, group1: 8 SGPRs) and issues tensor_load_to_lds.
// LDS padding: pad_amount=4 dwords every K dwords -> row pitch K+4 floats,
// which de-conflicts the 16 row-parallel LDS banks in the A-fragment reads.
// TDM zero-fills rows beyond tensor_dim1, so ragged M tiles need no clamping.

__device__ __forceinline__ void tdm_load_tile(const float* gsrc, unsigned lds_byte,
                                              int K, int rows_avail, int tile_rows,
                                              int pad_interval_code, int pad_amount_code) {
  unsigned long long ga = (unsigned long long)(uintptr_t)gsrc;
  u32x4 g0;
  g0[0] = 1u;                                            // count=1
  g0[1] = lds_byte;                                      // lds_addr [63:32]
  g0[2] = (unsigned)(ga & 0xFFFFFFFFu);                  // global_addr [95:64]
  g0[3] = (unsigned)((ga >> 32) & 0x01FFFFFFu) | (2u << 30);  // addr[56:32] | type=2
  u32x8 g1;
  unsigned ra = (unsigned)rows_avail;
  g1[0] = (2u << 16)                                     // data_size = 4B
        | (1u << 20)                                     // pad_enable
        | ((unsigned)pad_interval_code << 22)
        | ((unsigned)pad_amount_code << 25);
  g1[1] = ((unsigned)K & 0xFFFFu) << 16;                 // tensor_dim0[15:0]
  g1[2] = (((unsigned)K >> 16) & 0xFFFFu) | ((ra & 0xFFFFu) << 16);   // t_dim0[31:16] | t_dim1[15:0]
  g1[3] = ((ra >> 16) & 0xFFFFu) | (((unsigned)K & 0xFFFFu) << 16);   // t_dim1[31:16] | tile_dim0
  g1[4] = (unsigned)tile_rows & 0xFFFFu;                 // tile_dim1 (tile_dim2 = 0)
  g1[5] = (unsigned)K;                                   // tensor_dim0_stride[31:0]
  g1[6] = 0u;
  g1[7] = 0u;
  asm volatile("tensor_load_to_lds %0, %1" :: "s"(g0), "s"(g1) : "memory");
}

// ----------------------------------------------- WMMA bf16 GEMM, N = 128
// out(Mx128) = [scale .* A](MxK) @ Wbf^T + bias  [+ A2 @ W2bf^T]  [relu]
// One block = 16 output rows staged in LDS by TDM; 8 waves cover the 8 col tiles.

template <bool RELU, bool TWO, bool SCALE, int K>
__global__ __launch_bounds__(256) void wmma_gemm_kernel(
    const float* __restrict__ A, const float* __restrict__ scale,
    const __bf16* __restrict__ Wbf, const float* __restrict__ bias,
    const float* __restrict__ A2, const __bf16* __restrict__ W2bf,
    float* __restrict__ out, int M) {
  constexpr int PITCH = K + 4;                 // TDM-padded LDS row pitch (floats)
  constexpr int PI_CODE = (K == 128) ? 6 : 7;  // pad interval: 2^(code+1) dwords = K
  __shared__ float ldsA[(TWO ? 32 : 16) * PITCH];

  const int row0 = blockIdx.x * 16;
  const unsigned ldsBase = (unsigned)(uintptr_t)&ldsA[0];
  if (threadIdx.x < 32) {
    tdm_load_tile(A + (long long)row0 * K, ldsBase, K, M - row0, 16, PI_CODE, 3);
    if (TWO)
      tdm_load_tile(A2 + (long long)row0 * K, ldsBase + 16u * PITCH * 4u, K, M - row0, 16, PI_CODE, 3);
    __builtin_amdgcn_s_wait_tensorcnt(0);
  }
  __syncthreads();

  const int lane = threadIdx.x & 31;
  const int tileN = threadIdx.x >> 5;          // 0..7
  const int col0 = tileN * 16;

  // A-fragment: lane holds row (lane&15); 16 consecutive K at kb = (lane>=16)*16
  const int aRow = lane & 15;
  const int akb = (lane >> 4) * 16;
  // B-fragment: lane holds col (lane&15); 16 consecutive K at kb = (lane>=16)*16
  const int bn = col0 + (lane & 15);
  const int bkb = (lane >> 4) * 16;

  float s1 = 1.0f;
  if (SCALE) {
    int sr = row0 + aRow;
    s1 = scale[(sr < M) ? sr : (M - 1)];
  }

  v8f c = {};
  const float* pa0 = ldsA + aRow * PITCH + akb;
  const __bf16* wrow = Wbf + (long long)bn * K + bkb;
#pragma unroll
  for (int k0 = 0; k0 < K; k0 += 32) {
    v16bf a;
    const float* pa = pa0 + k0;
#pragma unroll
    for (int i = 0; i < 8; ++i) a[i] = (__bf16)(pa[i] * s1);
#pragma unroll
    for (int i = 0; i < 8; ++i) a[8 + i] = (__bf16)(pa[8 + i] * s1);
    v16bf b = *(const v16bf*)(wrow + k0);
    c = __builtin_amdgcn_wmma_f32_16x16x32_bf16(false, a, false, b, (short)0, c, false, false);
  }
  if (TWO) {
    const float* pa20 = ldsA + (16 + aRow) * PITCH + akb;
    const __bf16* w2row = W2bf + (long long)bn * K + bkb;
#pragma unroll
    for (int k0 = 0; k0 < K; k0 += 32) {
      v16bf a;
      const float* pa = pa20 + k0;
#pragma unroll
      for (int i = 0; i < 8; ++i) a[i] = (__bf16)pa[i];
#pragma unroll
      for (int i = 0; i < 8; ++i) a[8 + i] = (__bf16)pa[8 + i];
      v16bf b = *(const v16bf*)(w2row + k0);
      c = __builtin_amdgcn_wmma_f32_16x16x32_bf16(false, a, false, b, (short)0, c, false, false);
    }
  }

  // C/D: lane element r -> row row0 + (lane<16 ? r : 8+r), col bn
  const float bi = bias[bn];
  const int mb = (lane >> 4) * 8;
  if (row0 + 16 <= M) {                         // fast path: unguarded stores
#pragma unroll
    for (int r = 0; r < 8; ++r) {
      float v = c[r] + bi;
      if (RELU) v = fmaxf(v, 0.0f);
      out[(long long)(row0 + mb + r) * HFEAT + bn] = v;
    }
  } else {
#pragma unroll
    for (int r = 0; r < 8; ++r) {
      int rr = row0 + mb + r;
      if (rr < M) {
        float v = c[r] + bi;
        if (RELU) v = fmaxf(v, 0.0f);
        out[(long long)rr * HFEAT + bn] = v;
      }
    }
  }
}

// ------------------------------------------------- final H->1 projection

__global__ void out_kernel(const float* __restrict__ h, const float* __restrict__ W,
                           const float* __restrict__ b, float* __restrict__ out, int N) {
  int warp = (blockIdx.x * blockDim.x + threadIdx.x) >> 5;
  int lane = threadIdx.x & 31;
  int nwarp = (gridDim.x * blockDim.x) >> 5;
  for (int n = warp; n < N; n += nwarp) {
    float4 x = ((const float4*)(h + (long long)n * HFEAT))[lane];
    float4 w = ((const float4*)W)[lane];
    float s = x.x * w.x + x.y * w.y + x.z * w.z + x.w * w.w;
#pragma unroll
    for (int off = 16; off > 0; off >>= 1) s += __shfl_xor(s, off, 32);
    if (lane == 0) out[n] = s + b[0];
  }
}

// ---------------------------------------------------------------- driver

static inline int blks(long long threads, int tb) {
  return (int)((threads + tb - 1) / tb);
}

extern "C" void kernel_launch(void* const* d_in, const int* in_sizes, int n_in,
                              void* d_out, int out_size, void* d_ws, size_t ws_size,
                              hipStream_t stream) {
  // setup_inputs() dict order
  const float* net_x    = (const float*)d_in[0];
  const float* pin_in_x = (const float*)d_in[1];
  const float* pin_out_x= (const float*)d_in[2];
  const float* cell_x   = (const float*)d_in[3];
  const int* src_np = (const int*)d_in[4];
  const int* dst_np = (const int*)d_in[5];
  const int* src_pc = (const int*)d_in[6];
  const int* dst_pc = (const int*)d_in[7];
  const int* src_cp = (const int*)d_in[8];
  const int* dst_cp = (const int*)d_in[9];
  const int* src_pn = (const int*)d_in[10];
  const int* dst_pn = (const int*)d_in[11];
  const float* W_net = (const float*)d_in[12];
  const float* b_net = (const float*)d_in[13];
  const float* W_pi  = (const float*)d_in[14];
  const float* b_pi  = (const float*)d_in[15];
  const float* W_po  = (const float*)d_in[16];
  const float* b_po  = (const float*)d_in[17];
  const float* W_cs  = (const float*)d_in[18];
  const float* b_cs  = (const float*)d_in[19];
  const float* W_ct  = (const float*)d_in[20];
  const float* b_ct  = (const float*)d_in[21];
  const float* W_m1  = (const float*)d_in[22];
  const float* b_m1  = (const float*)d_in[23];
  const float* W_m2  = (const float*)d_in[24];
  const float* b_m2  = (const float*)d_in[25];
  const float* W_l   = (const float*)d_in[26];
  const float* b_l   = (const float*)d_in[27];
  const float* W_r   = (const float*)d_in[28];
  const float* W_out = (const float*)d_in[29];
  const float* b_out = (const float*)d_in[30];

  const int N_net  = in_sizes[0] / 4;
  const int N_pi   = in_sizes[1] / 3;
  const int N_po   = in_sizes[2] / 3;
  const int N_cell = in_sizes[3] / 32;
  const int E_np = in_sizes[4];
  const int E_pc = in_sizes[6];
  const int E_cp = in_sizes[8];
  const int E_pn = in_sizes[10];
  const long long szWl = (long long)in_sizes[26];   // 4*4*128*128
  const long long szWr = (long long)in_sizes[28];
  const long long szM1 = (long long)in_sizes[22];   // 128*256
  const long long szM2 = (long long)in_sizes[24];   // 128*128

  // workspace carve-up (f32 region, then bf16 region)
  float* ws = (float*)d_ws;
  long long off = 0;
  auto carve = [&](long long n) { float* p = ws + off; off += n; return p; };
  float* net_h  = carve((long long)N_net  * HFEAT);
  float* pi_h   = carve((long long)N_pi   * HFEAT);
  float* po_h   = carve((long long)N_po   * HFEAT);
  float* cell_h = carve((long long)N_cell * HFEAT);
  float* new_net  = carve((long long)N_net  * HFEAT);
  float* new_pi   = carve((long long)N_pi   * HFEAT);
  float* new_po   = carve((long long)N_po   * HFEAT);
  float* new_cell = carve((long long)N_cell * HFEAT);
  long long accN = (long long)N_pi * HFEAT;
  if ((long long)N_po * HFEAT > accN) accN = (long long)N_po * HFEAT;
  if ((long long)N_cell * 256 > accN) accN = (long long)N_cell * 256;
  if ((long long)N_net * HFEAT > accN) accN = (long long)N_net * HFEAT;
  float* acc = carve(accN);
  float* inv_pi   = carve(N_pi);
  float* inv_cell = carve(N_cell);
  float* inv_po   = carve(N_po);
  float* inv_net  = carve(N_net);
  long long invTotal = (long long)N_pi + N_cell + N_po + N_net;
  // bf16 weights (alignment: all counts are multiples of 128 floats)
  __bf16* bfBase = (__bf16*)(ws + off);
  long long boff = 0;
  auto carveB = [&](long long n) { __bf16* p = bfBase + boff; boff += n; return p; };
  __bf16* Wl_bf  = carveB(szWl);
  __bf16* Wr_bf  = carveB(szWr);
  __bf16* Wm1_bf = carveB(szM1);
  __bf16* Wm2_bf = carveB(szM2);

  const int TB = 256;

  // 0) bf16 weight conversion (once per launch; deterministic)
  cvt_bf16_kernel<<<blks(szWl, TB), TB, 0, stream>>>(W_l,  Wl_bf,  szWl);
  cvt_bf16_kernel<<<blks(szWr, TB), TB, 0, stream>>>(W_r,  Wr_bf,  szWr);
  cvt_bf16_kernel<<<blks(szM1, TB), TB, 0, stream>>>(W_m1, Wm1_bf, szM1);
  cvt_bf16_kernel<<<blks(szM2, TB), TB, 0, stream>>>(W_m2, Wm2_bf, szM2);

  // 1) degree counts -> 1/max(cnt,1)   (constant across layers)
  zero_f32_kernel<<<blks(invTotal, TB), TB, 0, stream>>>(inv_pi, invTotal);
  count_kernel<<<blks(E_np, TB), TB, 0, stream>>>(dst_np, inv_pi, E_np);
  count_kernel<<<blks(E_pc, TB), TB, 0, stream>>>(dst_pc, inv_cell, E_pc);
  count_kernel<<<blks(E_cp, TB), TB, 0, stream>>>(dst_cp, inv_po, E_cp);
  count_kernel<<<blks(E_pn, TB), TB, 0, stream>>>(dst_pn, inv_net, E_pn);
  inv_kernel<<<blks(invTotal, TB), TB, 0, stream>>>(inv_pi, invTotal);

  // 2) encoders
  encode_small_kernel<<<blks((long long)N_net * HFEAT, TB), TB, 0, stream>>>(
      net_x, W_net, b_net, net_h, N_net, 4, 4, 0, HFEAT, 0);
  encode_small_kernel<<<blks((long long)N_pi * HFEAT, TB), TB, 0, stream>>>(
      pin_in_x, W_pi, b_pi, pi_h, N_pi, 3, 3, 0, HFEAT, 0);
  encode_small_kernel<<<blks((long long)N_po * HFEAT, TB), TB, 0, stream>>>(
      pin_out_x, W_po, b_po, po_h, N_po, 3, 3, 0, HFEAT, 0);
  encode_small_kernel<<<blks((long long)N_cell * HFEAT, TB), TB, 0, stream>>>(
      cell_x, W_cs, b_cs, acc, N_cell, 6, 32, 26, 256, 0);
  encode_small_kernel<<<blks((long long)N_cell * HFEAT, TB), TB, 0, stream>>>(
      cell_x, W_ct, b_ct, acc, N_cell, 26, 32, 0, 256, HFEAT);
  wmma_gemm_kernel<true, false, false, 256><<<(N_cell + 15) / 16, 256, 0, stream>>>(
      acc, nullptr, Wm1_bf, b_m1, nullptr, nullptr, new_cell, N_cell);
  wmma_gemm_kernel<false, false, false, 128><<<(N_cell + 15) / 16, 256, 0, stream>>>(
      new_cell, nullptr, Wm2_bf, b_m2, nullptr, nullptr, cell_h, N_cell);

  // 3) message-passing layers
  for (int l = 0; l < 4; ++l) {
    const __bf16* Wl0 = Wl_bf + (long long)(l * 4 + 0) * HFEAT * HFEAT;
    const __bf16* Wl1 = Wl_bf + (long long)(l * 4 + 1) * HFEAT * HFEAT;
    const __bf16* Wl2 = Wl_bf + (long long)(l * 4 + 2) * HFEAT * HFEAT;
    const __bf16* Wl3 = Wl_bf + (long long)(l * 4 + 3) * HFEAT * HFEAT;
    const __bf16* Wr0 = Wr_bf + (long long)(l * 4 + 0) * HFEAT * HFEAT;
    const __bf16* Wr1 = Wr_bf + (long long)(l * 4 + 1) * HFEAT * HFEAT;
    const __bf16* Wr2 = Wr_bf + (long long)(l * 4 + 2) * HFEAT * HFEAT;
    const __bf16* Wr3 = Wr_bf + (long long)(l * 4 + 3) * HFEAT * HFEAT;
    const float* bl0 = b_l + (long long)(l * 4 + 0) * HFEAT;
    const float* bl1 = b_l + (long long)(l * 4 + 1) * HFEAT;
    const float* bl2 = b_l + (long long)(l * 4 + 2) * HFEAT;
    const float* bl3 = b_l + (long long)(l * 4 + 3) * HFEAT;

    // conv0: net -> pin_in
    zero_f32_kernel<<<blks((long long)N_pi * HFEAT, TB), TB, 0, stream>>>(acc, (long long)N_pi * HFEAT);
    scatter_kernel<<<blks((long long)E_np * 32, TB), TB, 0, stream>>>(net_h, src_np, dst_np, acc, E_np);
    wmma_gemm_kernel<false, true, true, 128><<<(N_pi + 15) / 16, 256, 0, stream>>>(
        acc, inv_pi, Wl0, bl0, pi_h, Wr0, new_pi, N_pi);

    // conv1: pin_in -> cell
    zero_f32_kernel<<<blks((long long)N_cell * HFEAT, TB), TB, 0, stream>>>(acc, (long long)N_cell * HFEAT);
    scatter_kernel<<<blks((long long)E_pc * 32, TB), TB, 0, stream>>>(pi_h, src_pc, dst_pc, acc, E_pc);
    wmma_gemm_kernel<false, true, true, 128><<<(N_cell + 15) / 16, 256, 0, stream>>>(
        acc, inv_cell, Wl1, bl1, cell_h, Wr1, new_cell, N_cell);

    // conv2: cell -> pin_out
    zero_f32_kernel<<<blks((long long)N_po * HFEAT, TB), TB, 0, stream>>>(acc, (long long)N_po * HFEAT);
    scatter_kernel<<<blks((long long)E_cp * 32, TB), TB, 0, stream>>>(cell_h, src_cp, dst_cp, acc, E_cp);
    wmma_gemm_kernel<false, true, true, 128><<<(N_po + 15) / 16, 256, 0, stream>>>(
        acc, inv_po, Wl2, bl2, po_h, Wr2, new_po, N_po);

    // conv3: pin_out -> net
    zero_f32_kernel<<<blks((long long)N_net * HFEAT, TB), TB, 0, stream>>>(acc, (long long)N_net * HFEAT);
    scatter_kernel<<<blks((long long)E_pn * 32, TB), TB, 0, stream>>>(po_h, src_pn, dst_pn, acc, E_pn);
    wmma_gemm_kernel<false, true, true, 128><<<(N_net + 15) / 16, 256, 0, stream>>>(
        acc, inv_net, Wl3, bl3, net_h, Wr3, new_net, N_net);

    // residual + relu updates (after all four convs, matching reference)
    update_kernel<<<blks((long long)N_pi * HFEAT, TB), TB, 0, stream>>>(pi_h, new_pi, (long long)N_pi * HFEAT);
    update_kernel<<<blks((long long)N_cell * HFEAT, TB), TB, 0, stream>>>(cell_h, new_cell, (long long)N_cell * HFEAT);
    update_kernel<<<blks((long long)N_po * HFEAT, TB), TB, 0, stream>>>(po_h, new_po, (long long)N_po * HFEAT);
    update_kernel<<<blks((long long)N_net * HFEAT, TB), TB, 0, stream>>>(net_h, new_net, (long long)N_net * HFEAT);
  }

  // 4) output projection
  out_kernel<<<blks((long long)N_net * 32, TB), TB, 0, stream>>>(
      net_h, W_out, b_out, (float*)d_out, N_net);
}
